// ParallelGroupedMLP_27255862460663
// MI455X (gfx1250) — compile-verified, matrix operand order checked by reading, not run on hardware
//
#include <hip/hip_runtime.h>
#include <math.h>

#define T_TOK  2048
#define H_DIM  1024
#define FF_DIM 4096
#define E_EXP  8
#define L_LORA 8
#define R_RANK 16
#define G_GRP  64

// GEMM tiling
#define BM 256
#define BN 128
#define BK 32
#define LDK 40   // padded K stride (ushorts): rows 80B apart -> 16B aligned, conflict-free frag reads

typedef __attribute__((ext_vector_type(16))) __bf16 v16bf;
typedef __attribute__((ext_vector_type(8)))  float  v8f;
typedef __attribute__((ext_vector_type(8)))  unsigned short ushort8;
typedef __attribute__((ext_vector_type(4)))  int    v4i;

#if __has_builtin(__builtin_amdgcn_global_load_async_to_lds_b128)
  #define HAVE_ASYNC_LDS 1
#else
  #define HAVE_ASYNC_LDS 0
#endif
#if __has_builtin(__builtin_amdgcn_s_wait_asynccnt)
  #define WAIT_ASYNC() __builtin_amdgcn_s_wait_asynccnt(0)
#else
  #define WAIT_ASYNC() asm volatile("s_wait_asynccnt 0" ::: "memory")
#endif

#define AS1 __attribute__((address_space(1)))
#define AS3 __attribute__((address_space(3)))

__device__ __forceinline__ AS3 v4i* lds_cast_v4(void* p) {
  // generic LDS pointer: low 32 bits are the LDS offset
  return (AS3 v4i*)(unsigned int)(unsigned long long)p;
}
__device__ __forceinline__ AS1 v4i* gbl_cast_v4(const void* p) {
  return (AS1 v4i*)(unsigned long long)p;
}

__device__ __forceinline__ unsigned short f2bf(float f) {
  unsigned int u = __float_as_uint(f);
  u += 0x7FFFu + ((u >> 16) & 1u);           // round-to-nearest-even
  return (unsigned short)(u >> 16);
}
__device__ __forceinline__ float bf2f(unsigned short b) {
  return __uint_as_float(((unsigned int)b) << 16);
}
__device__ __forceinline__ float gelu_f(float v) {
  const float k = 0.7978845608028654f;       // sqrt(2/pi)
  float u = k * (v + 0.044715f * v * v * v);
  return 0.5f * v * (1.0f + tanhf(u));
}

union FragU { v16bf f; ushort8 h[2]; };

// WMMA 16x16x32 bf16 fragment from LDS tile stored [row][K] with stride LDK (ushorts).
// lane<16: row=lane,    K halves {0..7},{16..23}
// lane>=16: row=lane-16, K halves {8..15},{24..31}
__device__ __forceinline__ v16bf ld_frag(const unsigned short* base, int lane) {
  int r  = lane & 15;
  int hi = (lane >> 4) << 3;                 // 0 or 8
  const unsigned short* p = base + r * LDK;
  FragU u;
  u.h[0] = *(const ushort8*)(p + hi);
  u.h[1] = *(const ushort8*)(p + 16 + hi);
  return u.f;
}

// A-tile staging: 256 rows x 32 bf16, thread covers rows tid>>2 + 64*i, 8-ushort segment (tid&3)*8.
__device__ __forceinline__ void issue_loadA(const unsigned short* src, size_t stride,
                                            unsigned short* dst, int am, int asg,
                                            ushort8* areg) {
#if HAVE_ASYNC_LDS
  #pragma unroll
  for (int i = 0; i < 4; ++i)
    __builtin_amdgcn_global_load_async_to_lds_b128(
        gbl_cast_v4(src + (size_t)(am + 64 * i) * stride + asg),
        lds_cast_v4(dst + (am + 64 * i) * LDK + asg), 0, 0);
  (void)areg;
#else
  #pragma unroll
  for (int i = 0; i < 4; ++i)
    areg[i] = *(const ushort8*)(src + (size_t)(am + 64 * i) * stride + asg);
  (void)dst;
#endif
}
__device__ __forceinline__ void commit_loadA(unsigned short* dst, int am, int asg,
                                             const ushort8* areg) {
#if !HAVE_ASYNC_LDS
  #pragma unroll
  for (int i = 0; i < 4; ++i)
    *(ushort8*)(dst + (am + 64 * i) * LDK + asg) = areg[i];
#else
  (void)dst; (void)am; (void)asg; (void)areg;
#endif
}

// ---------------- x -> bf16 pre-cast (kills 32x fp32 re-reads of x in GEMM1) ----------------
__global__ __launch_bounds__(256)
void xcast_kernel(const float* __restrict__ x, unsigned short* __restrict__ xb, int n) {
  for (int i = blockIdx.x * blockDim.x + threadIdx.x; i < n; i += gridDim.x * blockDim.x)
    xb[i] = f2bf(x[i]);
}

// ---------------- Router: per-expert linear -> softmax -> top2 ----------------
__global__ __launch_bounds__(256)
void router_topk_kernel(const float* __restrict__ x, const int* __restrict__ tpe,
                        const float* __restrict__ rw, int* __restrict__ ids,
                        float* __restrict__ wts) {
  int t = blockIdx.x * 8 + (threadIdx.x >> 5);   // one wave32 per token
  if (t >= T_TOK) return;
  int lane = threadIdx.x & 31;

  int e = E_EXP - 1; int end = 0;
  for (int i = 0; i < E_EXP; ++i) { end += tpe[i]; if (t < end) { e = i; break; } }

  float a[L_LORA];
  #pragma unroll
  for (int l = 0; l < L_LORA; ++l) a[l] = 0.0f;
  const float* xr = x + (size_t)t * H_DIM;
  const float* wr = rw + (size_t)e * H_DIM * L_LORA;
  for (int h = lane; h < H_DIM; h += 32) {
    float xv = xr[h];
    #pragma unroll
    for (int l = 0; l < L_LORA; ++l) a[l] += xv * wr[h * L_LORA + l];
  }
  #pragma unroll
  for (int off = 16; off >= 1; off >>= 1) {
    #pragma unroll
    for (int l = 0; l < L_LORA; ++l) a[l] += __shfl_xor(a[l], off, 32);
  }
  if (lane == 0) {
    float m = a[0];
    #pragma unroll
    for (int l = 1; l < L_LORA; ++l) m = fmaxf(m, a[l]);
    float p[L_LORA]; float s = 0.0f;
    #pragma unroll
    for (int l = 0; l < L_LORA; ++l) { p[l] = __expf(a[l] - m); s += p[l]; }
    float inv = 1.0f / s;
    int i0 = 0;
    #pragma unroll
    for (int l = 1; l < L_LORA; ++l) if (p[l] > p[i0]) i0 = l;
    int i1 = (i0 == 0) ? 1 : 0;
    #pragma unroll
    for (int l = 0; l < L_LORA; ++l) if (l != i0 && p[l] > p[i1]) i1 = l;
    ids[t * 2 + 0] = e * L_LORA + i0;  wts[t * 2 + 0] = p[i0] * inv;
    ids[t * 2 + 1] = e * L_LORA + i1;  wts[t * 2 + 1] = p[i1] * inv;
  }
}

// ---------------- GEMM1: x1 = x @ w1[e]^T  (w1: [E, FF, H], K=H contiguous) ----------------
// Double-buffered, software-pipelined: async A (bf16), register-staged B (fp32->bf16).
__global__ __launch_bounds__(256)
void gemm1_wmma_kernel(const unsigned short* __restrict__ xb, const float* __restrict__ w1,
                       const int* __restrict__ tpe, float* __restrict__ x1) {
  __shared__ unsigned short As[2][BM * LDK];
  __shared__ unsigned short Bs[2][BN * LDK];
  const int row0 = blockIdx.y * BM;
  const int col0 = blockIdx.x * BN;
  int e = E_EXP - 1; { int end = 0;
    for (int i = 0; i < E_EXP; ++i) { end += tpe[i]; if (row0 < end) { e = i; break; } } }
  const float* wb = w1 + (size_t)e * FF_DIM * H_DIM;
  const int tid = threadIdx.x, lane = tid & 31, wave = tid >> 5;
  const int wm = wave & 3;   // 4 M-waves x 64 rows
  const int wn = wave >> 2;  // 2 N-waves x 64 cols

  const int am  = tid >> 2, asg = (tid & 3) * 8;   // A staging coords
  const int bn  = tid >> 3, bk  = (tid & 7) * 4;   // B staging coords

  v8f c[4][4];
  #pragma unroll
  for (int i = 0; i < 4; ++i)
    #pragma unroll
    for (int j = 0; j < 4; ++j)
      #pragma unroll
      for (int v = 0; v < 8; ++v) c[i][j][v] = 0.0f;

  const unsigned short* arow = xb + (size_t)row0 * H_DIM;
  ushort8 areg[4];
  float4  breg[4];

  // prologue: stage k-block 0 into buffer 0
  issue_loadA(arow, H_DIM, &As[0][0], am, asg, areg);
  #pragma unroll
  for (int i = 0; i < 4; ++i)
    breg[i] = *(const float4*)(wb + (size_t)(col0 + bn + 32 * i) * H_DIM + bk);
  commit_loadA(&As[0][0], am, asg, areg);
  #pragma unroll
  for (int i = 0; i < 4; ++i) {
    unsigned short* d = &Bs[0][(bn + 32 * i) * LDK + bk];
    d[0] = f2bf(breg[i].x); d[1] = f2bf(breg[i].y);
    d[2] = f2bf(breg[i].z); d[3] = f2bf(breg[i].w);
  }
  WAIT_ASYNC();
  __syncthreads();

  const int nkb = H_DIM / BK;
  for (int kb = 0; kb < nkb; ++kb) {
    const int cur = kb & 1;
    const bool more = (kb + 1) < nkb;
    const int k1 = (kb + 1) * BK;
    if (more) {   // fire next tile's loads before computing current one
      issue_loadA(arow + k1, H_DIM, &As[cur ^ 1][0], am, asg, areg);
      #pragma unroll
      for (int i = 0; i < 4; ++i)
        breg[i] = *(const float4*)(wb + (size_t)(col0 + bn + 32 * i) * H_DIM + k1 + bk);
    }
    // compute current tile (overlaps with in-flight loads)
    v16bf af[4];
    #pragma unroll
    for (int i = 0; i < 4; ++i) af[i] = ld_frag(&As[cur][(wm * 64 + i * 16) * LDK], lane);
    #pragma unroll
    for (int j = 0; j < 4; ++j) {
      v16bf bfrag = ld_frag(&Bs[cur][(wn * 64 + j * 16) * LDK], lane);
      #pragma unroll
      for (int i = 0; i < 4; ++i)
        c[i][j] = __builtin_amdgcn_wmma_f32_16x16x32_bf16(false, af[i], false, bfrag,
                                                          (short)0, c[i][j], false, false);
    }
    if (more) {   // commit next tile into the other buffer
      const int nb = cur ^ 1;
      commit_loadA(&As[nb][0], am, asg, areg);
      #pragma unroll
      for (int i = 0; i < 4; ++i) {
        unsigned short* d = &Bs[nb][(bn + 32 * i) * LDK + bk];
        d[0] = f2bf(breg[i].x); d[1] = f2bf(breg[i].y);
        d[2] = f2bf(breg[i].z); d[3] = f2bf(breg[i].w);
      }
      WAIT_ASYNC();
    }
    __syncthreads();
  }

  const int rlo = lane >> 4, cn = lane & 15;
  #pragma unroll
  for (int i = 0; i < 4; ++i)
    #pragma unroll
    for (int j = 0; j < 4; ++j) {
      int colc = col0 + wn * 64 + j * 16 + cn;
      #pragma unroll
      for (int v = 0; v < 8; ++v) {
        int r = row0 + wm * 64 + i * 16 + v + 8 * rlo;
        x1[(size_t)r * FF_DIM + colc] = c[i][j][v];
      }
    }
}

// ---------------- GEMM2: out = h @ w2[e]  (w2: [E, FF, H], K=FF strided) ----------------
__global__ __launch_bounds__(256)
void gemm2_wmma_kernel(const unsigned short* __restrict__ hbf, const float* __restrict__ w2,
                       const int* __restrict__ tpe, float* __restrict__ out) {
  __shared__ unsigned short As[2][BM * LDK];
  __shared__ unsigned short Bs[2][BN * LDK];
  const int row0 = blockIdx.y * BM;
  const int col0 = blockIdx.x * BN;
  int e = E_EXP - 1; { int end = 0;
    for (int i = 0; i < E_EXP; ++i) { end += tpe[i]; if (row0 < end) { e = i; break; } } }
  const float* wb = w2 + (size_t)e * FF_DIM * H_DIM;
  const int tid = threadIdx.x, lane = tid & 31, wave = tid >> 5;
  const int wm = wave & 3, wn = wave >> 2;

  const int am  = tid >> 2, asg = (tid & 3) * 8;   // A staging coords
  const int bk2 = tid >> 5, bn2 = (tid & 31) * 4;  // B staging coords (transposing load)

  v8f c[4][4];
  #pragma unroll
  for (int i = 0; i < 4; ++i)
    #pragma unroll
    for (int j = 0; j < 4; ++j)
      #pragma unroll
      for (int v = 0; v < 8; ++v) c[i][j][v] = 0.0f;

  const unsigned short* arow = hbf + (size_t)row0 * FF_DIM;
  ushort8 areg[4];
  float4  breg[4];

  // prologue
  issue_loadA(arow, FF_DIM, &As[0][0], am, asg, areg);
  #pragma unroll
  for (int i = 0; i < 4; ++i)
    breg[i] = *(const float4*)(wb + (size_t)(bk2 + 8 * i) * H_DIM + col0 + bn2);
  commit_loadA(&As[0][0], am, asg, areg);
  #pragma unroll
  for (int i = 0; i < 4; ++i) {
    int k = bk2 + 8 * i;
    Bs[0][(bn2 + 0) * LDK + k] = f2bf(breg[i].x);
    Bs[0][(bn2 + 1) * LDK + k] = f2bf(breg[i].y);
    Bs[0][(bn2 + 2) * LDK + k] = f2bf(breg[i].z);
    Bs[0][(bn2 + 3) * LDK + k] = f2bf(breg[i].w);
  }
  WAIT_ASYNC();
  __syncthreads();

  const int nkb = FF_DIM / BK;
  for (int kb = 0; kb < nkb; ++kb) {
    const int cur = kb & 1;
    const bool more = (kb + 1) < nkb;
    const int k1 = (kb + 1) * BK;
    if (more) {
      issue_loadA(arow + k1, FF_DIM, &As[cur ^ 1][0], am, asg, areg);
      #pragma unroll
      for (int i = 0; i < 4; ++i)
        breg[i] = *(const float4*)(wb + (size_t)(k1 + bk2 + 8 * i) * H_DIM + col0 + bn2);
    }
    v16bf af[4];
    #pragma unroll
    for (int i = 0; i < 4; ++i) af[i] = ld_frag(&As[cur][(wm * 64 + i * 16) * LDK], lane);
    #pragma unroll
    for (int j = 0; j < 4; ++j) {
      v16bf bfrag = ld_frag(&Bs[cur][(wn * 64 + j * 16) * LDK], lane);
      #pragma unroll
      for (int i = 0; i < 4; ++i)
        c[i][j] = __builtin_amdgcn_wmma_f32_16x16x32_bf16(false, af[i], false, bfrag,
                                                          (short)0, c[i][j], false, false);
    }
    if (more) {
      const int nb = cur ^ 1;
      commit_loadA(&As[nb][0], am, asg, areg);
      #pragma unroll
      for (int i = 0; i < 4; ++i) {
        int k = bk2 + 8 * i;
        Bs[nb][(bn2 + 0) * LDK + k] = f2bf(breg[i].x);
        Bs[nb][(bn2 + 1) * LDK + k] = f2bf(breg[i].y);
        Bs[nb][(bn2 + 2) * LDK + k] = f2bf(breg[i].z);
        Bs[nb][(bn2 + 3) * LDK + k] = f2bf(breg[i].w);
      }
      WAIT_ASYNC();
    }
    __syncthreads();
  }

  const int rlo = lane >> 4, cn = lane & 15;
  #pragma unroll
  for (int i = 0; i < 4; ++i)
    #pragma unroll
    for (int j = 0; j < 4; ++j) {
      int colc = col0 + wn * 64 + j * 16 + cn;
      #pragma unroll
      for (int v = 0; v < 8; ++v) {
        int r = row0 + wm * 64 + i * 16 + v + 8 * rlo;
        out[(size_t)r * H_DIM + colc] = c[i][j][v];
      }
    }
}

// ---------------- LoRA path 1: x1[t] += w0*(x[t]@A0@B0) + w1*(x[t]@A1@B1) ----------------
__global__ __launch_bounds__(256)
void lora1_kernel(const float* __restrict__ x, const int* __restrict__ ids,
                  const float* __restrict__ wts, const float* __restrict__ A,
                  const float* __restrict__ B, float* __restrict__ x1) {
  __shared__ float part[256];
  __shared__ float u_sm[2][R_RANK];
  const int t = blockIdx.x, tid = threadIdx.x;
  const int g0 = ids[t * 2], g1 = ids[t * 2 + 1];
  const float w0 = wts[t * 2], w1 = wts[t * 2 + 1];

  const int kk = tid >> 7, r = (tid >> 3) & 15, p = tid & 7;
  const int g = kk ? g1 : g0;
  const float* Ab = A + (size_t)g * H_DIM * R_RANK;
  const float* xr = x + (size_t)t * H_DIM;
  float s = 0.0f;
  for (int h = p; h < H_DIM; h += 8) s += xr[h] * Ab[h * R_RANK + r];
  part[tid] = s;
  __syncthreads();
  if (tid < 32) {
    float acc = 0.0f;
    #pragma unroll
    for (int q = 0; q < 8; ++q) acc += part[tid * 8 + q];
    u_sm[tid >> 4][tid & 15] = acc;
  }
  __syncthreads();
  const float* B0 = B + (size_t)g0 * R_RANK * FF_DIM;
  const float* B1 = B + (size_t)g1 * R_RANK * FF_DIM;
  for (int f = tid; f < FF_DIM; f += 256) {
    float s0 = 0.0f, s1 = 0.0f;
    #pragma unroll
    for (int rr = 0; rr < R_RANK; ++rr) {
      s0 += u_sm[0][rr] * B0[rr * FF_DIM + f];
      s1 += u_sm[1][rr] * B1[rr * FF_DIM + f];
    }
    x1[(size_t)t * FF_DIM + f] += w0 * s0 + w1 * s1;
  }
}

// ---------------- GELU + bf16 cast: h = bf16(gelu(x1)) ----------------
__global__ __launch_bounds__(256)
void gelu_bf16_kernel(const float* __restrict__ x1, unsigned short* __restrict__ hbf, int n) {
  for (int i = blockIdx.x * blockDim.x + threadIdx.x; i < n; i += gridDim.x * blockDim.x)
    hbf[i] = f2bf(gelu_f(x1[i]));
}

// ---------------- LoRA path 2: out[t] += w0*(h[t]@A0@B0) + w1*(h[t]@A1@B1) ----------------
__global__ __launch_bounds__(256)
void lora2_kernel(const unsigned short* __restrict__ hbf, const int* __restrict__ ids,
                  const float* __restrict__ wts, const float* __restrict__ A,
                  const float* __restrict__ B, float* __restrict__ out) {
  __shared__ float part[256];
  __shared__ float u_sm[2][R_RANK];
  const int t = blockIdx.x, tid = threadIdx.x;
  const int g0 = ids[t * 2], g1 = ids[t * 2 + 1];
  const float w0 = wts[t * 2], w1 = wts[t * 2 + 1];

  const int kk = tid >> 7, r = (tid >> 3) & 15, p = tid & 7;
  const int g = kk ? g1 : g0;
  const float* Ab = A + (size_t)g * FF_DIM * R_RANK;
  const unsigned short* hr = hbf + (size_t)t * FF_DIM;
  float s = 0.0f;
  for (int h = p; h < FF_DIM; h += 8) s += bf2f(hr[h]) * Ab[h * R_RANK + r];
  part[tid] = s;
  __syncthreads();
  if (tid < 32) {
    float acc = 0.0f;
    #pragma unroll
    for (int q = 0; q < 8; ++q) acc += part[tid * 8 + q];
    u_sm[tid >> 4][tid & 15] = acc;
  }
  __syncthreads();
  const float* B0 = B + (size_t)g0 * R_RANK * H_DIM;
  const float* B1 = B + (size_t)g1 * R_RANK * H_DIM;
  for (int o = tid; o < H_DIM; o += 256) {
    float s0 = 0.0f, s1 = 0.0f;
    #pragma unroll
    for (int rr = 0; rr < R_RANK; ++rr) {
      s0 += u_sm[0][rr] * B0[rr * H_DIM + o];
      s1 += u_sm[1][rr] * B1[rr * H_DIM + o];
    }
    out[(size_t)t * H_DIM + o] += w0 * s0 + w1 * s1;
  }
}

extern "C" void kernel_launch(void* const* d_in, const int* in_sizes, int n_in,
                              void* d_out, int out_size, void* d_ws, size_t ws_size,
                              hipStream_t stream) {
  (void)in_sizes; (void)n_in; (void)out_size; (void)ws_size;
  const float* x   = (const float*)d_in[0];
  const int*   tpe = (const int*)d_in[1];
  const float* rw  = (const float*)d_in[2];
  const float* w1  = (const float*)d_in[3];
  const float* w2  = (const float*)d_in[4];
  const float* w1A = (const float*)d_in[5];
  const float* w1B = (const float*)d_in[6];
  const float* w2A = (const float*)d_in[7];
  const float* w2B = (const float*)d_in[8];
  float* out = (float*)d_out;

  char* ws = (char*)d_ws;
  size_t off = 0;
  int* ids = (int*)(ws + off);
  off += ((size_t)T_TOK * 2 * sizeof(int) + 255) & ~(size_t)255;
  float* wts = (float*)(ws + off);
  off += ((size_t)T_TOK * 2 * sizeof(float) + 255) & ~(size_t)255;
  float* x1 = (float*)(ws + off);
  off += (size_t)T_TOK * FF_DIM * sizeof(float);
  unsigned short* hbf = (unsigned short*)(ws + off);
  off += (size_t)T_TOK * FF_DIM * sizeof(unsigned short);
  unsigned short* xb = (unsigned short*)(ws + off);
  off += (size_t)T_TOK * H_DIM * sizeof(unsigned short);

  xcast_kernel<<<2048, 256, 0, stream>>>(x, xb, T_TOK * H_DIM);
  router_topk_kernel<<<T_TOK / 8, 256, 0, stream>>>(x, tpe, rw, ids, wts);
  gemm1_wmma_kernel<<<dim3(FF_DIM / BN, T_TOK / BM), 256, 0, stream>>>(xb, w1, tpe, x1);
  lora1_kernel<<<T_TOK, 256, 0, stream>>>(x, ids, wts, w1A, w1B, x1);
  gelu_bf16_kernel<<<4096, 256, 0, stream>>>(x1, hbf, T_TOK * FF_DIM);
  gemm2_wmma_kernel<<<dim3(H_DIM / BN, T_TOK / BM), 256, 0, stream>>>(hbf, w2, tpe, out);
  lora2_kernel<<<T_TOK, 256, 0, stream>>>(hbf, ids, wts, w2A, w2B, out);
}
